// SolubilityGNN_67250597921402
// MI455X (gfx1250) — matrix operand order repaired and checked by reading, not compiled, not run on hardware
//
#include <hip/hip_runtime.h>
#include <hip/hip_bf16.h>

typedef __attribute__((ext_vector_type(16))) __bf16 v16bf;
typedef __attribute__((ext_vector_type(8)))  __bf16 v8bf;
typedef __attribute__((ext_vector_type(4)))  __bf16 v4bf;
typedef __attribute__((ext_vector_type(8)))  float  v8f;

#define HDIM 128
#define EPS 1e-5f

// ---------------------------------------------------------------------------
// Utility fills / conversions
// ---------------------------------------------------------------------------
__global__ void fill_f32(float* __restrict__ p, float val, long long n) {
    long long i = (long long)blockIdx.x * blockDim.x + threadIdx.x;
    if (i < n) p[i] = val;
}

__global__ void f32_to_bf16_k(const float* __restrict__ in, __bf16* __restrict__ out, long long n) {
    long long i = (long long)blockIdx.x * blockDim.x + threadIdx.x;
    if (i < n) out[i] = (__bf16)in[i];
}

// Wt[n][k] = (bf16) W[k][n]   (W is 128x128 row-major, k-major)
__global__ void w_transpose_bf16_k(const float* __restrict__ W, __bf16* __restrict__ Wt) {
    int idx = blockIdx.x * blockDim.x + threadIdx.x;   // 0 .. 16383
    int n = idx >> 7;
    int k = idx & 127;
    Wt[idx] = (__bf16)W[k * HDIM + n];
}

// ---------------------------------------------------------------------------
// Degree / normalization:  deg[i] = 1 (self loop) + #edges with dst==i
// ---------------------------------------------------------------------------
__global__ void deg_edges_k(const int* __restrict__ dst, float* __restrict__ deg, long long E) {
    long long e = (long long)blockIdx.x * blockDim.x + threadIdx.x;
    if (e < E) atomicAdd(&deg[dst[e]], 1.0f);
}

__global__ void deg_to_dinv_k(float* __restrict__ deg, long long n) {
    long long i = (long long)blockIdx.x * blockDim.x + threadIdx.x;
    if (i < n) deg[i] = rsqrtf(deg[i]);
}

// ---------------------------------------------------------------------------
// WMMA GEMM:  C[M x 128] (bf16 out) = A[M x 128] (bf16, row-major) * Wt^T
// Wt stored transposed: Wt[n][k].  One wave -> one 16x16 tile; block = 8 waves
// = 16 rows x 128 cols.  M is a multiple of 16 (grid = M/16) -> no guards.
// ---------------------------------------------------------------------------
__global__ void gemm_bf16_wmma_k(const __bf16* __restrict__ A,
                                 const __bf16* __restrict__ Bt,
                                 __bf16* __restrict__ C) {
    const int lane = threadIdx.x & 31;
    const int wave = threadIdx.x >> 5;           // 0..7 -> column tile
    const int half = lane >> 4;                  // 0/1 lane half
    const int r16  = lane & 15;
    const int m0   = blockIdx.x * 16;
    const int n0   = wave * 16;

    const int arow = m0 + r16;
    const int brow = n0 + r16;
    const int aoff = half * 16;                  // A: 16 contiguous K per lane-half
    const int boff = half * 16;                  // B: 16 contiguous K per lane-half

    v8f acc = {};
#pragma unroll
    for (int kb = 0; kb < HDIM; kb += 32) {
        const __bf16* pa = A + (size_t)arow * HDIM + kb;
        v8bf alo = *(const v8bf*)(pa + aoff);
        v8bf ahi = *(const v8bf*)(pa + 8 + aoff);
        v16bf a;
#pragma unroll
        for (int i = 0; i < 8; ++i) { a[i] = alo[i]; a[i + 8] = ahi[i]; }

        v16bf b = *(const v16bf*)(Bt + (size_t)brow * HDIM + kb + boff);

        // 8 args: (neg_a, A, neg_b, B, c_mod, C, reuse_a, reuse_b)
        acc = __builtin_amdgcn_wmma_f32_16x16x32_bf16(
            false, a, false, b, (short)0, acc, false, false);
    }

#pragma unroll
    for (int v = 0; v < 8; ++v) {
        const int row = m0 + v + half * 8;
        C[(size_t)row * HDIM + n0 + r16] = (__bf16)acc[v];
    }
}

// ---------------------------------------------------------------------------
// Aggregation: agg = self-loop term (init, no pre-zero needed), then edge scatter
// ---------------------------------------------------------------------------
__global__ void self_loop_init_k(const __bf16* __restrict__ hlin,
                                 const float* __restrict__ dinv,
                                 float* __restrict__ agg, long long n) {
    long long i = (long long)blockIdx.x * blockDim.x + threadIdx.x;  // over N*128
    if (i < n) {
        long long node = i >> 7;
        float d = dinv[node];
        agg[i] = (float)hlin[i] * d * d;
    }
}

// one wave (32 lanes) per edge; each lane handles 4 features (bf16 gather, f32 atomics)
__global__ void edge_scatter_k(const __bf16* __restrict__ hlin,
                               float* __restrict__ agg,
                               const int* __restrict__ src,
                               const int* __restrict__ dst,
                               const float* __restrict__ dinv, long long E) {
    long long tid = (long long)blockIdx.x * blockDim.x + threadIdx.x;
    long long e = tid >> 5;
    int lane = (int)(tid & 31);
    if (e >= E) return;
    const int s = src[e];
    const int d = dst[e];
    const float norm = dinv[s] * dinv[d];
    const v4bf hv = *(const v4bf*)(hlin + (size_t)s * HDIM + lane * 4);
    float* po = agg + (size_t)d * HDIM + lane * 4;
    atomicAdd(po + 0, (float)hv[0] * norm);
    atomicAdd(po + 1, (float)hv[1] * norm);
    atomicAdd(po + 2, (float)hv[2] * norm);
    atomicAdd(po + 3, (float)hv[3] * norm);
}

// bias + BN(eval) + ReLU; writes bf16 (next GEMM input) and optionally f32.
// out_f32 may alias agg (same-index read-then-write per thread).
__global__ void epilogue_k(const float* __restrict__ agg,
                           const float* __restrict__ bias,
                           const float* __restrict__ g, const float* __restrict__ be,
                           const float* __restrict__ m, const float* __restrict__ v,
                           __bf16* __restrict__ out_bf, float* out_f32,
                           long long n) {
    long long i = (long long)blockIdx.x * blockDim.x + threadIdx.x;
    if (i < n) {
        int h = (int)(i & 127);
        float y = agg[i] + bias[h];
        y = (y - m[h]) * rsqrtf(v[h] + EPS) * g[h] + be[h];
        y = fmaxf(y, 0.0f);
        out_bf[i] = (__bf16)y;
        if (out_f32) out_f32[i] = y;
    }
}

// ---------------------------------------------------------------------------
// Global mean pool + FC head
// ---------------------------------------------------------------------------
__global__ void pool_scatter_k(const float* __restrict__ h3,
                               const int* __restrict__ batch,
                               float* __restrict__ pooled, long long Nn) {
    long long tid = (long long)blockIdx.x * blockDim.x + threadIdx.x;
    long long node = tid >> 5;
    int lane = (int)(tid & 31);
    if (node >= Nn) return;
    const int gidx = batch[node];
    const float4 hv = *(const float4*)(h3 + (size_t)node * HDIM + lane * 4);
    float* po = pooled + (size_t)gidx * HDIM + lane * 4;
    atomicAdd(po + 0, hv.x);
    atomicAdd(po + 1, hv.y);
    atomicAdd(po + 2, hv.z);
    atomicAdd(po + 3, hv.w);
}

__global__ void pool_count_k(const int* __restrict__ batch, float* __restrict__ cnt, long long Nn) {
    long long i = (long long)blockIdx.x * blockDim.x + threadIdx.x;
    if (i < Nn) atomicAdd(&cnt[batch[i]], 1.0f);
}

__global__ void pool_fc_k(const float* __restrict__ pooled, const float* __restrict__ cnt,
                          const float* __restrict__ fw1, const float* __restrict__ fb1,
                          const float* __restrict__ fw2, const float* __restrict__ fb2,
                          float* __restrict__ out) {
    __shared__ float sp[HDIM];
    __shared__ float sz[32];
    const int gidx = blockIdx.x;
    const int t = threadIdx.x;
    sp[t] = pooled[(size_t)gidx * HDIM + t] / fmaxf(cnt[gidx], 1.0f);
    __syncthreads();
    if (t < 32) {
        float z = fb1[t];
#pragma unroll 8
        for (int h = 0; h < HDIM; ++h) z += sp[h] * fw1[h * 32 + t];
        sz[t] = fmaxf(z, 0.0f);
    }
    __syncthreads();
    if (t == 0) {
        float o = fb2[0];
#pragma unroll
        for (int j = 0; j < 32; ++j) o += sz[j] * fw2[j];
        out[gidx] = o;
    }
}

// ---------------------------------------------------------------------------
// Launch
// ---------------------------------------------------------------------------
static inline size_t align256(size_t x) { return (x + 255) & ~(size_t)255; }

extern "C" void kernel_launch(void* const* d_in, const int* in_sizes, int n_in,
                              void* d_out, int out_size, void* d_ws, size_t ws_size,
                              hipStream_t stream) {
    const long long N = in_sizes[0] / HDIM;     // 50000
    const long long E = in_sizes[1] / 2;        // 800000
    const long long G = out_size;               // 2000

    const float* x    = (const float*)d_in[0];
    const int*   src  = (const int*)d_in[1];
    const int*   dst  = (const int*)d_in[1] + E;
    const int*   batch= (const int*)d_in[2];
    const float* W[3] = {(const float*)d_in[3], (const float*)d_in[5], (const float*)d_in[7]};
    const float* B[3] = {(const float*)d_in[4], (const float*)d_in[6], (const float*)d_in[8]};
    const float* fw1  = (const float*)d_in[9];
    const float* fb1  = (const float*)d_in[10];
    const float* fw2  = (const float*)d_in[11];
    const float* fb2  = (const float*)d_in[12];
    const float* BNg[3] = {(const float*)d_in[13], (const float*)d_in[17], (const float*)d_in[21]};
    const float* BNb[3] = {(const float*)d_in[14], (const float*)d_in[18], (const float*)d_in[22]};
    const float* BNm[3] = {(const float*)d_in[15], (const float*)d_in[19], (const float*)d_in[23]};
    const float* BNv[3] = {(const float*)d_in[16], (const float*)d_in[20], (const float*)d_in[24]};
    float* out = (float*)d_out;

    // ---- workspace layout ----
    char* ws = (char*)d_ws;
    size_t off = 0;
    __bf16* xb   = (__bf16*)(ws + off); off += align256((size_t)N * HDIM * sizeof(__bf16));
    __bf16* wt   = (__bf16*)(ws + off); off += align256((size_t)HDIM * HDIM * sizeof(__bf16));
    __bf16* hlin = (__bf16*)(ws + off); off += align256((size_t)N * HDIM * sizeof(__bf16));
    float* agg   = (float*)(ws + off);  off += align256((size_t)N * HDIM * sizeof(float));
    float* dinv  = (float*)(ws + off);  off += align256((size_t)N * sizeof(float));
    float* pooled= (float*)(ws + off);  off += align256((size_t)G * HDIM * sizeof(float));
    float* cnt   = (float*)(ws + off);  off += align256((size_t)G * sizeof(float));
    float* h3    = agg;   // layer-3 epilogue writes fp32 in place over agg (same index)

    const int T = 256;
    const long long NH = N * HDIM;
    dim3 blkNH((unsigned)((NH + T - 1) / T));
    dim3 blkE((unsigned)((E + T - 1) / T));
    dim3 blkE32((unsigned)((E * 32 + T - 1) / T));
    dim3 blkN32((unsigned)((N * 32 + T - 1) / T));
    dim3 blkN((unsigned)((N + T - 1) / T));

    // ---- degree / dinv (shared by all layers) ----
    fill_f32<<<blkN, T, 0, stream>>>(dinv, 1.0f, N);
    deg_edges_k<<<blkE, T, 0, stream>>>(dst, dinv, E);
    deg_to_dinv_k<<<blkN, T, 0, stream>>>(dinv, N);

    // ---- input features -> bf16 ----
    f32_to_bf16_k<<<blkNH, T, 0, stream>>>(x, xb, NH);

    // ---- three GCN layers ----
    const unsigned gemm_blocks = (unsigned)(N / 16);          // 3125 (N multiple of 16)
    for (int L = 0; L < 3; ++L) {
        w_transpose_bf16_k<<<(HDIM * HDIM) / T, T, 0, stream>>>(W[L], wt);
        gemm_bf16_wmma_k<<<gemm_blocks, T, 0, stream>>>(xb, wt, hlin);
        self_loop_init_k<<<blkNH, T, 0, stream>>>(hlin, dinv, agg, NH);
        edge_scatter_k<<<blkE32, T, 0, stream>>>(hlin, agg, src, dst, dinv, E);
        epilogue_k<<<blkNH, T, 0, stream>>>(agg, B[L], BNg[L], BNb[L], BNm[L], BNv[L],
                                            xb, (L == 2) ? h3 : (float*)nullptr, NH);
    }

    // ---- global mean pool + FC head ----
    fill_f32<<<(unsigned)((G * HDIM + T - 1) / T), T, 0, stream>>>(pooled, 0.0f, G * HDIM);
    fill_f32<<<(unsigned)((G + T - 1) / T), T, 0, stream>>>(cnt, 0.0f, G);
    pool_scatter_k<<<blkN32, T, 0, stream>>>(h3, batch, pooled, N);
    pool_count_k<<<blkN, T, 0, stream>>>(batch, cnt, N);
    pool_fc_k<<<(unsigned)G, HDIM, 0, stream>>>(pooled, cnt, fw1, fb1, fw2, fb2, out);
}
